// gs_block_17497696764589
// MI455X (gfx1250) — matrix operand order repaired and checked by reading, not compiled, not run on hardware
//
#include <hip/hip_runtime.h>
#include <hip/hip_bf16.h>

// ---------------------------------------------------------------------------
// GraphSAGE-style GCN block on MI455X (gfx1250, wave32, WMMA).
//   neigh = (adj @ x) / rowsum(adj)        GEMM1: f16 WMMA, f32 accum
//   h     = relu(neigh @ W^T)              GEMM2: f16 WMMA, f32 accum
//   out   = h / max(||h||_2, 1e-12)        row normalize
// adj is binary -> exact in f16. deg fused into GEMM1's adj pass.
// ---------------------------------------------------------------------------

#define N_ROWS  8192
#define FEAT    512
#define EMBED   512

typedef __attribute__((ext_vector_type(16))) _Float16 v16h;
typedef __attribute__((ext_vector_type(8)))  float    v8f;

union AU { v16h v; _Float16 e[16]; };
union BU { float4 f4[2]; v16h v; };

// ---------------------------------------------------------------------------
// Prep: zero accumulator arrays (deg, ss) -- needed every call (atomics).
// ---------------------------------------------------------------------------
__global__ __launch_bounds__(256) void k_zero(float* __restrict__ deg,
                                              float* __restrict__ ss) {
    int i = blockIdx.x * blockDim.x + threadIdx.x;
    if (i < N_ROWS) { deg[i] = 0.0f; ss[i] = 0.0f; }
}

// ---------------------------------------------------------------------------
// Pack x [8192,512] f32 -> f16 in B-fragment-native order.
// Tile = (K=32 rows of x) x (N=16 cols). tile id t = kb*32 + nb.
// Lane L of tile t holds N = nb*16 + (L&15), K = kb*32 + 8*(L>>4) + h (h<8),
// K = kb*32 + 16 + 8*(L>>4) + (h-8) (h>=8),
// stored as 16 contiguous halves (32 B) at xf[(t*32+L)*16 + h].
// ---------------------------------------------------------------------------
__global__ __launch_bounds__(256) void k_pack_x(const float* __restrict__ x,
                                                _Float16* __restrict__ xf) {
    int gid  = blockIdx.x * blockDim.x + threadIdx.x;   // 0 .. 262143
    int lane = gid & 31;
    int t    = gid >> 5;           // 0 .. 8191
    int kb   = t >> 5;             // 0 .. 255
    int nb   = t & 31;             // 0 .. 31
    int jbase = kb * 32 + ((lane >> 4) << 3);
    int f     = nb * 16 + (lane & 15);
    _Float16* dst = xf + ((size_t)t * 32 + lane) * 16;
#pragma unroll
    for (int h = 0; h < 16; ++h)
        dst[h] = (_Float16)x[(size_t)(jbase + h + (h & 8)) * FEAT + f];
}

// ---------------------------------------------------------------------------
// Pack W [512,512] (row-major [e,f]) as B = W^T (K=f, N=e), same fragment order.
// ---------------------------------------------------------------------------
__global__ __launch_bounds__(256) void k_pack_w(const float* __restrict__ W,
                                                _Float16* __restrict__ wf) {
    int gid  = blockIdx.x * blockDim.x + threadIdx.x;   // 0 .. 16383
    int lane = gid & 31;
    int t    = gid >> 5;           // 0 .. 511
    int kb   = t >> 5;             // 0 .. 15
    int nb   = t & 31;             // 0 .. 31
    int fbase = kb * 32 + ((lane >> 4) << 3);
    int e     = nb * 16 + (lane & 15);
    _Float16* dst = wf + ((size_t)t * 32 + lane) * 16;
#pragma unroll
    for (int h = 0; h < 16; ++h)
        dst[h] = (_Float16)W[(size_t)e * FEAT + fbase + h + (h & 8)];
}

// ---------------------------------------------------------------------------
// GEMM1: raw = adj @ x  (f16 WMMA, f32 accum) + fused deg = rowsum(adj).
// Grid (64, 4): block tile M=128, N=128. 8 waves: wave w owns rows
// [m0, m0+16). Each wave: 8 accumulators (16x16 each) covering N=128.
// A fragment (16x32 f16) built per the ISA layout:
//   lane L: M = L&15; halves 0-7  -> K = 8*(L>>4) + h
//                     halves 8-15 -> K = 16 + 8*(L>>4) + (h-8)
// ---------------------------------------------------------------------------
__global__ __launch_bounds__(256) void k_gemm1(const float* __restrict__ adj,
                                               const _Float16* __restrict__ xf,
                                               float* __restrict__ raw,
                                               float* __restrict__ deg) {
    const int lane = threadIdx.x & 31;
    const int wave = threadIdx.x >> 5;
    const int m0   = blockIdx.x * 128 + wave * 16;
    const int row  = m0 + (lane & 15);
    const int kg   = (lane >> 4) * 8;
    const float* __restrict__ arow = adj + (size_t)row * N_ROWS;

    v8f acc[8];
#pragma unroll
    for (int i = 0; i < 8; ++i) acc[i] = (v8f){0,0,0,0,0,0,0,0};

    float degsum = 0.0f;

    for (int kb = 0; kb < N_ROWS / 32; ++kb) {
        const int k0 = kb * 32;
        // A: 16 f32 per lane (4x b128), convert to f16 (adj is 0/1 -> exact)
        const float4 a0 = *(const float4*)(arow + k0 + kg);
        const float4 a1 = *(const float4*)(arow + k0 + kg + 4);
        const float4 a2 = *(const float4*)(arow + k0 + kg + 16);
        const float4 a3 = *(const float4*)(arow + k0 + kg + 20);

        degsum += (a0.x + a0.y + a0.z + a0.w) + (a1.x + a1.y + a1.z + a1.w)
                + (a2.x + a2.y + a2.z + a2.w) + (a3.x + a3.y + a3.z + a3.w);

        AU au;
        au.e[0]  = (_Float16)a0.x; au.e[1]  = (_Float16)a0.y;
        au.e[2]  = (_Float16)a0.z; au.e[3]  = (_Float16)a0.w;
        au.e[4]  = (_Float16)a1.x; au.e[5]  = (_Float16)a1.y;
        au.e[6]  = (_Float16)a1.z; au.e[7]  = (_Float16)a1.w;
        au.e[8]  = (_Float16)a2.x; au.e[9]  = (_Float16)a2.y;
        au.e[10] = (_Float16)a2.z; au.e[11] = (_Float16)a2.w;
        au.e[12] = (_Float16)a3.x; au.e[13] = (_Float16)a3.y;
        au.e[14] = (_Float16)a3.z; au.e[15] = (_Float16)a3.w;

        // stream-prefetch adj two K-steps ahead (global_prefetch_b8)
        if (kb + 2 < N_ROWS / 32)
            __builtin_prefetch(arow + k0 + 64 + kg, 0, 1);

#pragma unroll
        for (int nb = 0; nb < 8; ++nb) {
            const size_t t = (size_t)kb * 32 + blockIdx.y * 8 + nb;
            BU bu;
            const float4* bp = (const float4*)(xf + (t * 32 + lane) * 16);
            bu.f4[0] = bp[0];
            bu.f4[1] = bp[1];
            acc[nb] = __builtin_amdgcn_wmma_f32_16x16x32_f16(
                false, au.v, false, bu.v, (short)0, acc[nb], false, false);
        }
    }

    // fused degree: lanes L and L+16 cover complementary K subsets of one row
    degsum += __shfl_xor(degsum, 16, 32);
    if (blockIdx.y == 0 && lane < 16)
        atomicAdd(&deg[row], degsum);

    // C layout: VGPR r -> row (m0 + 8*(L>>4) + r), col = n0 + (L&15)
    const int col0  = blockIdx.y * 128 + (lane & 15);
    const int rbase = m0 + ((lane >> 4) << 3);
#pragma unroll
    for (int nb = 0; nb < 8; ++nb) {
#pragma unroll
        for (int r = 0; r < 8; ++r)
            raw[(size_t)(rbase + r) * FEAT + col0 + nb * 16] = acc[nb][r];
    }
}

// ---------------------------------------------------------------------------
// GEMM2: h = relu((raw/deg) @ W^T), store h to out, fused ss = rowsum(h*h).
// Same tiling as GEMM1; K = 512 (16 steps).
// ---------------------------------------------------------------------------
__global__ __launch_bounds__(256) void k_gemm2(const float* __restrict__ raw,
                                               const float* __restrict__ deg,
                                               const _Float16* __restrict__ wf,
                                               float* __restrict__ out,
                                               float* __restrict__ ss) {
    const int lane = threadIdx.x & 31;
    const int wave = threadIdx.x >> 5;
    const int m0   = blockIdx.x * 128 + wave * 16;
    const int row  = m0 + (lane & 15);
    const int kg   = (lane >> 4) * 8;
    const float rdeg = 1.0f / deg[row];   // deg >= 1 (self loops)
    const float* __restrict__ rrow = raw + (size_t)row * FEAT;

    v8f acc[8];
#pragma unroll
    for (int i = 0; i < 8; ++i) acc[i] = (v8f){0,0,0,0,0,0,0,0};

    for (int kb = 0; kb < FEAT / 32; ++kb) {
        const int k0 = kb * 32;
        const float4 a0 = *(const float4*)(rrow + k0 + kg);
        const float4 a1 = *(const float4*)(rrow + k0 + kg + 4);
        const float4 a2 = *(const float4*)(rrow + k0 + kg + 16);
        const float4 a3 = *(const float4*)(rrow + k0 + kg + 20);

        AU au;
        au.e[0]  = (_Float16)(a0.x * rdeg); au.e[1]  = (_Float16)(a0.y * rdeg);
        au.e[2]  = (_Float16)(a0.z * rdeg); au.e[3]  = (_Float16)(a0.w * rdeg);
        au.e[4]  = (_Float16)(a1.x * rdeg); au.e[5]  = (_Float16)(a1.y * rdeg);
        au.e[6]  = (_Float16)(a1.z * rdeg); au.e[7]  = (_Float16)(a1.w * rdeg);
        au.e[8]  = (_Float16)(a2.x * rdeg); au.e[9]  = (_Float16)(a2.y * rdeg);
        au.e[10] = (_Float16)(a2.z * rdeg); au.e[11] = (_Float16)(a2.w * rdeg);
        au.e[12] = (_Float16)(a3.x * rdeg); au.e[13] = (_Float16)(a3.y * rdeg);
        au.e[14] = (_Float16)(a3.z * rdeg); au.e[15] = (_Float16)(a3.w * rdeg);

#pragma unroll
        for (int nb = 0; nb < 8; ++nb) {
            const size_t t = (size_t)kb * 32 + blockIdx.y * 8 + nb;
            BU bu;
            const float4* bp = (const float4*)(wf + (t * 32 + lane) * 16);
            bu.f4[0] = bp[0];
            bu.f4[1] = bp[1];
            acc[nb] = __builtin_amdgcn_wmma_f32_16x16x32_f16(
                false, au.v, false, bu.v, (short)0, acc[nb], false, false);
        }
    }

    // epilogue: relu, store, fused per-row sum of squares
    const int col0  = blockIdx.y * 128 + (lane & 15);
    const int rbase = m0 + ((lane >> 4) << 3);
    float ssr[8];
#pragma unroll
    for (int r = 0; r < 8; ++r) ssr[r] = 0.0f;

#pragma unroll
    for (int nb = 0; nb < 8; ++nb) {
#pragma unroll
        for (int r = 0; r < 8; ++r) {
            float v = fmaxf(acc[nb][r], 0.0f);
            out[(size_t)(rbase + r) * EMBED + col0 + nb * 16] = v;
            ssr[r] += v * v;
        }
    }
    // reduce across the 16 lanes that share a row (stays within half-wave)
#pragma unroll
    for (int r = 0; r < 8; ++r) {
        float s = ssr[r];
        s += __shfl_xor(s, 1, 32);
        s += __shfl_xor(s, 2, 32);
        s += __shfl_xor(s, 4, 32);
        s += __shfl_xor(s, 8, 32);
        if ((lane & 15) == 0)
            atomicAdd(&ss[rbase + r], s);
    }
}

// ---------------------------------------------------------------------------
// Row L2-normalize in place: out[i,:] *= 1/max(sqrt(ss[i]), 1e-12).
// One block per row; 256 threads x 2 elements.
// ---------------------------------------------------------------------------
__global__ __launch_bounds__(256) void k_norm(float* __restrict__ out,
                                              const float* __restrict__ ss) {
    const int row = blockIdx.x;
    const float scale = 1.0f / fmaxf(sqrtf(ss[row]), 1e-12f);
    const size_t base = (size_t)row * EMBED + threadIdx.x;
    out[base]       *= scale;
    out[base + 256] *= scale;
}

// ---------------------------------------------------------------------------
extern "C" void kernel_launch(void* const* d_in, const int* in_sizes, int n_in,
                              void* d_out, int out_size, void* d_ws, size_t ws_size,
                              hipStream_t stream) {
    const float* x   = (const float*)d_in[0];   // [8192, 512]
    const float* adj = (const float*)d_in[1];   // [8192, 8192]
    const float* W   = (const float*)d_in[2];   // [512, 512]
    float*       out = (float*)d_out;           // [8192, 512]

    // workspace layout
    char* ws = (char*)d_ws;
    _Float16* xf  = (_Float16*)(ws);                               //  8 MB
    _Float16* wfb = (_Float16*)(ws + (size_t)8 * 1024 * 1024);     // 512 KB
    float*    raw = (float*)   (ws + (size_t)8704 * 1024);         // 16 MB
    float*    deg = (float*)   (ws + (size_t)25088 * 1024);        // 32 KB
    float*    ssb = (float*)   (ws + (size_t)25120 * 1024);        // 32 KB

    k_zero  <<<dim3(N_ROWS / 256), dim3(256), 0, stream>>>(deg, ssb);
    k_pack_x<<<dim3(1024),         dim3(256), 0, stream>>>(x, xf);
    k_pack_w<<<dim3(64),           dim3(256), 0, stream>>>(W, wfb);

    k_gemm1<<<dim3(N_ROWS / 128, FEAT / 128),  dim3(256), 0, stream>>>(adj, xf, raw, deg);
    k_gemm2<<<dim3(N_ROWS / 128, EMBED / 128), dim3(256), 0, stream>>>(raw, deg, wfb, out, ssb);
    k_norm <<<dim3(N_ROWS),                    dim3(256), 0, stream>>>(out, ssb);
}
